// PriorNetwork_9809705304291
// MI455X (gfx1250) — matrix operand order repaired and checked by reading, not compiled, not run on hardware
//
#include <hip/hip_runtime.h>
#include <hip/hip_bf16.h>
#include <math.h>
#include <stdint.h>

typedef __attribute__((ext_vector_type(16))) _Float16 v16h;
typedef __attribute__((ext_vector_type(8)))  _Float16 v8h;
typedef __attribute__((ext_vector_type(8)))  float    v8f;
typedef __attribute__((ext_vector_type(4)))  float    v4f;
typedef __attribute__((ext_vector_type(2)))  float    v2f;

#define NQ      2048
#define NT      100000
#define NTP     100096      // padded to a multiple of 64 rows (pad rows can't win)
#define D       256
#define NH      512
#define EPSV    1.0e-4f

// NN tiling: stage = 4 N-tiles (64 rows, 32KB) double-buffered in LDS,
// chunk = 17 stages (1088 rows), 92 chunks cover NTP exactly.
#define STAGE_TILES 4
#define STAGE_ROWS  64
#define ROW_PITCH   528              // 512B row + 16B pad (bank-quad friendly)
#define STAGE_BYTES (STAGE_ROWS * ROW_PITCH)   // 33792
#define STAGES      17
#define CHUNK_ROWS  (STAGES * STAGE_ROWS)      // 1088
#define NCHUNKS     92                          // 92*1088 == 100096 == NTP

union V16 { v16h v; v8h h[2]; };

// ---------------------------------------------------------------------------
// Phase A: f32 -> f16 conversion of codes + codes_table (+zero pad rows),
// and c2[n] = ||row||^2 (pad rows get 3e38 so they never win the argmin).
// One wave per 256-element row; 8 elements per lane.  Grid covers rows exactly.
// ---------------------------------------------------------------------------
__global__ __launch_bounds__(256) void cvt_kernel(const float* __restrict__ codes,
                                                  const float* __restrict__ table,
                                                  _Float16* __restrict__ qf16,
                                                  _Float16* __restrict__ tf16,
                                                  float* __restrict__ c2) {
    const int wave = threadIdx.x >> 5;
    const int lane = threadIdx.x & 31;
    const long long row = (long long)blockIdx.x * 8 + wave;   // < NTP + NQ exactly

    if (row < NT || row >= NTP) {     // uniform per wave
        const bool isTable = row < NT;
        const float* src = isTable ? (table + row * (long long)D)
                                   : (codes + (row - NTP) * (long long)D);
        _Float16*    dst = isTable ? (tf16 + row * (long long)D)
                                   : (qf16 + (row - NTP) * (long long)D);
        const v4f* s4 = (const v4f*)(src + lane * 8);
        v4f x0 = s4[0];
        v4f x1 = s4[1];
        v8h h;
        float ss = 0.f;
#pragma unroll
        for (int j = 0; j < 4; ++j) {
            h[j]     = (_Float16)x0[j];
            h[j + 4] = (_Float16)x1[j];
            ss = fmaf(x0[j], x0[j], ss);
            ss = fmaf(x1[j], x1[j], ss);
        }
        *(v8h*)(dst + lane * 8) = h;
        if (isTable) {
#pragma unroll
            for (int off = 16; off >= 1; off >>= 1)
                ss += __shfl_xor(ss, off);
            if (lane == 0) c2[row] = ss;
        }
    } else {                          // pad row: zero data, huge c2
        v8h z = {};
        *(v8h*)(tf16 + row * (long long)D + lane * 8) = z;
        if (lane == 0) c2[row] = 3.0e38f;
    }
}

// ---------------------------------------------------------------------------
// Phase B: nearest-neighbor search.
// Block = 4 waves; each wave holds 2 resident A m-tiles (32 queries, 128 VGPR).
// B tiles staged into LDS once per block with global_load_async_to_lds_b128
// (ASYNCcnt), double-buffered; every B fetched from LDS feeds 2 WMMAs.
// score = c2[n] - 2*dot == d2 - q2 (argmin-invariant).
// ---------------------------------------------------------------------------
__global__ __launch_bounds__(128) void nn_kernel(const _Float16* __restrict__ qf16,
                                                 const _Float16* __restrict__ tf16,
                                                 const float* __restrict__ c2,
                                                 float* __restrict__ pScore,
                                                 int*   __restrict__ pIdx) {
    __shared__ __align__(16) char lds[2 * STAGE_BYTES];

    const int tid    = threadIdx.x;
    const int wave   = tid >> 5;
    const int lane   = tid & 31;
    const int laneLo = lane & 15;
    const int hi     = lane >> 4;
    const int chunk  = blockIdx.y;
    const int rowChunk = chunk * CHUNK_ROWS;
    const int m0     = blockIdx.x * 128 + wave * 32;   // wave: m-tiles m0, m0+16

    // A operand resident: 2 m-tiles x K=256.  16-bit A 16x32 layout: lanes
    // 0-15 hold K 0..7 & 16..23, lanes 16-31 hold K 8..15 & 24..31.
    v16h a[2][8];
#pragma unroll
    for (int mt = 0; mt < 2; ++mt) {
        const _Float16* qrow = qf16 + (size_t)(m0 + mt * 16 + laneLo) * D;
#pragma unroll
        for (int kt = 0; kt < 8; ++kt) {
            V16 t;
            t.h[0] = *(const v8h*)(qrow + kt * 32 + hi * 8);
            t.h[1] = *(const v8h*)(qrow + kt * 32 + 16 + hi * 8);
            a[mt][kt] = t.v;
        }
    }

    float bestS[2][8];
    int   bestI[2][8];
#pragma unroll
    for (int mt = 0; mt < 2; ++mt)
#pragma unroll
        for (int v = 0; v < 8; ++v) { bestS[mt][v] = 3.4e38f; bestI[mt][v] = 0; }

    const uint32_t ldsBase = (uint32_t)(uintptr_t)(&lds[0]);

    // Cooperative async stage copy: 2048 x 16B transfers (32KB), 16 per thread,
    // rows padded to ROW_PITCH in LDS.  Consecutive lanes -> consecutive 16B.
    auto issue_stage = [&](int s, int buf) {
        const char* src = (const char*)(tf16 + (size_t)(rowChunk + s * STAGE_ROWS) * D);
        const uint32_t dstBase = ldsBase + (uint32_t)(buf * STAGE_BYTES);
#pragma unroll
        for (int i = 0; i < 16; ++i) {
            const int e = tid + 128 * i;
            const uint32_t ldsoff = dstBase + (uint32_t)(e * 16 + (e >> 5) * 16);
            const uint64_t gaddr  = (uint64_t)(uintptr_t)src + (uint64_t)(e * 16);
            asm volatile("global_load_async_to_lds_b128 %0, %1, off"
                         :: "v"(ldsoff), "v"(gaddr) : "memory");
        }
    };

    issue_stage(0, 0);
    for (int s = 0; s < STAGES; ++s) {
        asm volatile("s_wait_asynccnt 0x0" ::: "memory");
        __syncthreads();                       // stage s visible to all waves
        if (s + 1 < STAGES) issue_stage(s + 1, (s + 1) & 1);

        const char* buf = lds + (s & 1) * STAGE_BYTES;
#pragma unroll
        for (int t = 0; t < STAGE_TILES; ++t) {
            // B 32x16 layout: lane's column N = laneLo, K hi*8..+7 & 16+hi*8..+7.
            const char* rowp = buf + (t * 16 + laneLo) * ROW_PITCH + hi * 16;
            v8f acc0 = {};
            v8f acc1 = {};
#pragma unroll
            for (int kt = 0; kt < 8; ++kt) {
                V16 b;
                b.h[0] = *(const v8h*)(rowp + kt * 64);
                b.h[1] = *(const v8h*)(rowp + kt * 64 + 32);
                acc0 = __builtin_amdgcn_wmma_f32_16x16x32_f16(
                    false, a[0][kt], false, b.v, (short)0, acc0, false, false);
                acc1 = __builtin_amdgcn_wmma_f32_16x16x32_f16(
                    false, a[1][kt], false, b.v, (short)0, acc1, false, false);
            }
            const int   n   = rowChunk + s * STAGE_ROWS + t * 16 + laneLo;
            const float c2n = c2[n];
#pragma unroll
            for (int v = 0; v < 8; ++v) {
                float s0 = fmaf(-2.0f, acc0[v], c2n);
                float s1 = fmaf(-2.0f, acc1[v], c2n);
                bool  l0 = (s0 < bestS[0][v]);
                bool  l1 = (s1 < bestS[1][v]);
                bestS[0][v] = l0 ? s0 : bestS[0][v];
                bestI[0][v] = l0 ? n  : bestI[0][v];
                bestS[1][v] = l1 ? s1 : bestS[1][v];
                bestI[1][v] = l1 ? n  : bestI[1][v];
            }
        }
    }

    // min-reduce over the 16 lanes of each half (xor masks < 16 stay in-half)
#pragma unroll
    for (int mt = 0; mt < 2; ++mt) {
#pragma unroll
        for (int v = 0; v < 8; ++v) {
#pragma unroll
            for (int off = 1; off < 16; off <<= 1) {
                float os = __shfl_xor(bestS[mt][v], off);
                int   oi = __shfl_xor(bestI[mt][v], off);
                if (os < bestS[mt][v] || (os == bestS[mt][v] && oi < bestI[mt][v])) {
                    bestS[mt][v] = os; bestI[mt][v] = oi;
                }
            }
            if (laneLo == 0) {
                const int q = m0 + mt * 16 + v + hi * 8;
                pScore[(size_t)q * NCHUNKS + chunk] = bestS[mt][v];
                pIdx  [(size_t)q * NCHUNKS + chunk] = bestI[mt][v];
            }
        }
    }
}

// ---------------------------------------------------------------------------
// Reduce partial argmins over chunks -> nbr[q]
// ---------------------------------------------------------------------------
__global__ __launch_bounds__(256) void argmin_kernel(const float* __restrict__ pScore,
                                                     const int* __restrict__ pIdx,
                                                     int* __restrict__ nbr) {
    const int q = blockIdx.x * blockDim.x + threadIdx.x;
    if (q >= NQ) return;
    float bs = 3.4e38f;
    int   bi = 0;
    for (int c = 0; c < NCHUNKS; ++c) {
        const float s = pScore[(size_t)q * NCHUNKS + c];
        const int   i = pIdx  [(size_t)q * NCHUNKS + c];
        if (s < bs || (s == bs && i < bi)) { bs = s; bi = i; }
    }
    nbr[q] = bi;
}

// ---------------------------------------------------------------------------
// Phase C1: h1 = relu(prev_code @ w1^T + b1), f32 WMMA 16x16x4.
// ---------------------------------------------------------------------------
__global__ __launch_bounds__(32) void mlp1_kernel(const float* __restrict__ table,
                                                  const int* __restrict__ nbr,
                                                  const float* __restrict__ w1,
                                                  const float* __restrict__ b1,
                                                  float* __restrict__ h1) {
    const int lane   = threadIdx.x & 31;
    const int laneLo = lane & 15;
    const int hi     = lane >> 4;
    const int m0     = blockIdx.x * 16;
    const int n0     = blockIdx.y * 16;

    const int ridx = nbr[m0 + laneLo];
    const float* ap = table + (size_t)ridx * D;
    const float* bp = w1 + (size_t)(n0 + laneLo) * D;

    v8f acc = {};
    for (int k = 0; k < D; k += 4) {
        v2f av = *(const v2f*)(ap + k + hi * 2);
        v2f bv = *(const v2f*)(bp + k + hi * 2);
        acc = __builtin_amdgcn_wmma_f32_16x16x4_f32(
            false, av, false, bv, (short)0, acc, false, false);
    }
    const float bias = b1[n0 + laneLo];
#pragma unroll
    for (int v = 0; v < 8; ++v) {
        float h = acc[v] + bias;
        h = h > 0.f ? h : 0.f;
        h1[(size_t)(m0 + v + hi * 8) * NH + (n0 + laneLo)] = h;
    }
}

// ---------------------------------------------------------------------------
// Phase C2: mu = h1 @ w2u^T + b2u ; std = softplus(h1 @ w2s^T + b2s) + EPS
// ---------------------------------------------------------------------------
__global__ __launch_bounds__(32) void mlp2_kernel(const float* __restrict__ h1,
                                                  const float* __restrict__ w2u,
                                                  const float* __restrict__ b2u,
                                                  const float* __restrict__ w2s,
                                                  const float* __restrict__ b2s,
                                                  float* __restrict__ out) {
    const int lane   = threadIdx.x & 31;
    const int laneLo = lane & 15;
    const int hi     = lane >> 4;
    const int m0     = blockIdx.x * 16;
    const int n0     = blockIdx.y * 16;

    const float* ap  = h1  + (size_t)(m0 + laneLo) * NH;
    const float* bup = w2u + (size_t)(n0 + laneLo) * NH;
    const float* bsp = w2s + (size_t)(n0 + laneLo) * NH;

    v8f accU = {};
    v8f accS = {};
    for (int k = 0; k < NH; k += 4) {
        v2f av = *(const v2f*)(ap  + k + hi * 2);
        v2f bu = *(const v2f*)(bup + k + hi * 2);
        v2f bs = *(const v2f*)(bsp + k + hi * 2);
        accU = __builtin_amdgcn_wmma_f32_16x16x4_f32(
            false, av, false, bu, (short)0, accU, false, false);
        accS = __builtin_amdgcn_wmma_f32_16x16x4_f32(
            false, av, false, bs, (short)0, accS, false, false);
    }
    const float biasU = b2u[n0 + laneLo];
    const float biasS = b2s[n0 + laneLo];
#pragma unroll
    for (int v = 0; v < 8; ++v) {
        const int m = m0 + v + hi * 8;
        const int n = n0 + laneLo;
        out[(size_t)m * D + n] = accU[v] + biasU;
        float x  = accS[v] + biasS;
        float sp = (x > 20.f) ? x : log1pf(expf(x));
        out[(size_t)NQ * D + (size_t)m * D + n] = sp + EPSV;
    }
}

// ---------------------------------------------------------------------------
extern "C" void kernel_launch(void* const* d_in, const int* in_sizes, int n_in,
                              void* d_out, int out_size, void* d_ws, size_t ws_size,
                              hipStream_t stream) {
    const float* codes = (const float*)d_in[0];
    const float* table = (const float*)d_in[1];
    const float* w1    = (const float*)d_in[2];
    const float* b1    = (const float*)d_in[3];
    const float* w2u   = (const float*)d_in[4];
    const float* b2u   = (const float*)d_in[5];
    const float* w2s   = (const float*)d_in[6];
    const float* b2s   = (const float*)d_in[7];
    float* out = (float*)d_out;

    char* ws = (char*)d_ws;
    size_t off = 0;
    auto alloc = [&](size_t bytes) -> char* {
        char* p = ws + off;
        off = (off + bytes + 255) & ~(size_t)255;
        return p;
    };
    _Float16* tf16 = (_Float16*)alloc((size_t)NTP * D * sizeof(_Float16));
    _Float16* qf16 = (_Float16*)alloc((size_t)NQ * D * sizeof(_Float16));
    float*    c2   = (float*)   alloc((size_t)NTP * sizeof(float));
    float*    pScr = (float*)   alloc((size_t)NQ * NCHUNKS * sizeof(float));
    int*      pIdx = (int*)     alloc((size_t)NQ * NCHUNKS * sizeof(int));
    int*      nbr  = (int*)     alloc((size_t)NQ * sizeof(int));
    float*    h1   = (float*)   alloc((size_t)NQ * NH * sizeof(float));
    (void)ws_size; (void)in_sizes; (void)n_in; (void)out_size;

    cvt_kernel<<<dim3((NTP + NQ) / 8), 256, 0, stream>>>(codes, table, qf16, tf16, c2);
    nn_kernel<<<dim3(NQ / 128, NCHUNKS), 128, 0, stream>>>(qf16, tf16, c2, pScr, pIdx);
    argmin_kernel<<<dim3((NQ + 255) / 256), 256, 0, stream>>>(pScr, pIdx, nbr);
    mlp1_kernel<<<dim3(NQ / 16, NH / 16), 32, 0, stream>>>(table, nbr, w1, b1, h1);
    mlp2_kernel<<<dim3(NQ / 16, D / 16), 32, 0, stream>>>(h1, w2u, b2u, w2s, b2s, out);
}
